// GraphNetBlock_14345190768739
// MI455X (gfx1250) — matrix operand order repaired and checked by reading, not compiled, not run on hardware
//
#include <hip/hip_runtime.h>
#include <hip/hip_bf16.h>

#define ROWS 128
#define AS   136   // padded LDS stride in bf16 elements (272 B = 17*16, keeps 16B alignment)

typedef __attribute__((ext_vector_type(16))) __bf16 v16bf;
typedef __attribute__((ext_vector_type(8)))  float  v8f;
typedef unsigned int u32x4 __attribute__((ext_vector_type(4)));
typedef int          i32x4 __attribute__((ext_vector_type(4)));
typedef int          i32x8 __attribute__((ext_vector_type(8)));
typedef int v4i __attribute__((__vector_size__(4 * sizeof(int))));
typedef __attribute__((address_space(1))) v4i as1_v4i;
typedef __attribute__((address_space(3))) v4i as3_v4i;

#if defined(__has_builtin)
#if __has_builtin(__builtin_amdgcn_tensor_load_to_lds) && __has_builtin(__builtin_amdgcn_s_wait_tensorcnt)
#define HAS_TDM 1
#endif
#if __has_builtin(__builtin_amdgcn_global_load_async_to_lds_b128) && __has_builtin(__builtin_amdgcn_s_wait_asynccnt)
#define HAS_ASYNC 1
#endif
#endif

union Frag {
  v16bf v;
  uint4 q[2];
};

__device__ __forceinline__ unsigned short f2bf(float f) {
  unsigned int u = __float_as_uint(f);
  u += 0x7FFFu + ((u >> 16) & 1u);   // round-to-nearest-even
  return (unsigned short)(u >> 16);
}

__device__ __forceinline__ void zero_acc(v8f acc[8]) {
#pragma unroll
  for (int t = 0; t < 8; ++t)
#pragma unroll
    for (int i = 0; i < 8; ++i) acc[t][i] = 0.0f;
}

#ifdef HAS_ASYNC
__device__ __forceinline__ void async_copy16(const void* g, void* l) {
  __builtin_amdgcn_global_load_async_to_lds_b128(
      (as1_v4i*)(unsigned long long)(uintptr_t)g,
      (as3_v4i*)(unsigned int)(uintptr_t)l, 0, 0);
}
#endif

// acc += A(16-row slab, K=128 bf16 LDS tile) x W (W^T rows n-major in LDS)
__device__ __forceinline__ void gemm128(const unsigned short* __restrict__ A,
                                        const unsigned short* __restrict__ W,
                                        v8f acc[8], int wrow, int lane) {
  const int m     = lane & 15;
  const int koffA = (lane >> 4) * 8;    // A: lanes 16-31 hold K+8 / K+24 chunks
  const int koffB = (lane >> 4) * 8;    // B: lanes 16-31 hold K+8 / K+24 chunks
  const unsigned short* arow = A + (wrow + m) * AS;
#pragma unroll
  for (int k0 = 0; k0 < 128; k0 += 32) {
    Frag a;
    a.q[0] = *(const uint4*)(arow + k0 + koffA);
    a.q[1] = *(const uint4*)(arow + k0 + 16 + koffA);
#pragma unroll
    for (int t = 0; t < 8; ++t) {
      Frag b;
      const unsigned short* brow = W + (t * 16 + m) * AS + k0 + koffB;
      b.q[0] = *(const uint4*)(brow);
      b.q[1] = *(const uint4*)(brow + 16);
      acc[t] = __builtin_amdgcn_wmma_f32_16x16x32_bf16(
          false, a.v, false, b.v, (short)0, acc[t], false, false);
    }
  }
}

// bias + optional relu, convert to bf16, store wave's 16x128 slab into LDS tile
__device__ __forceinline__ void bias_act_store(v8f acc[8], const float* __restrict__ bias,
                                               unsigned short* __restrict__ dst,
                                               int wrow, int lane, bool relu) {
  const int m     = lane & 15;
  const int rbase = (lane >> 4) * 8;
#pragma unroll
  for (int t = 0; t < 8; ++t) {
    float bn = bias[t * 16 + m];
#pragma unroll
    for (int r = 0; r < 8; ++r) {
      float v = acc[t][r] + bn;
      if (relu) v = fmaxf(v, 0.0f);
      dst[(wrow + rbase + r) * AS + (t * 16 + m)] = f2bf(v);
    }
  }
}

// stage one 128x128 bf16 weight block (row stride ldw) into padded LDS tile.
// TDM path: single wave issues one tensor DMA descriptor; HW pads 16B per 256B row.
__device__ __forceinline__ void stage_w_any(const unsigned short* __restrict__ Wsrc, int ldw,
                                            unsigned short* __restrict__ sW, int tid) {
#ifdef HAS_TDM
  if (tid < 32) {
    unsigned long long ga = (unsigned long long)(uintptr_t)Wsrc;
    unsigned int lds = (unsigned int)(uintptr_t)sW;
    u32x4 g0;
    g0[0] = 1u;                                              // count=1, user mode
    g0[1] = lds;                                             // lds_addr
    g0[2] = (unsigned int)ga;                                // global_addr[31:0]
    g0[3] = (unsigned int)((ga >> 32) & 0x01FFFFFFu) | (2u << 30);  // addr[56:32] | type=2
    unsigned int s1 = (unsigned int)(ldw * 128);             // dim1 stride (unused for 2D tile)
    i32x8 g1;
    g1[0] = (int)((1u << 16) | (1u << 20) | (5u << 22) | (3u << 25)); // 2B elems, pad_en, 64DW/4DW
    g1[1] = (int)(((unsigned)ldw & 0xFFFFu) << 16);          // tensor_dim0 lo16 (<=384)
    g1[2] = (int)((128u & 0xFFFFu) << 16);                   // dim0 hi16=0 | tensor_dim1 lo16
    g1[3] = (int)(128u << 16);                               // dim1 hi16=0 | tile_dim0=128
    g1[4] = (int)128u;                                       // tile_dim1=128 | tile_dim2=0
    g1[5] = (int)(unsigned)ldw;                              // tensor_dim0_stride[31:0]
    g1[6] = (int)((s1 & 0xFFFFu) << 16);                     // stride0 hi16=0 | stride1 lo16
    g1[7] = (int)(s1 >> 16);                                 // stride1 upper bits
    i32x4 g2; g2[0] = 0; g2[1] = 0; g2[2] = 0; g2[3] = 0;
    i32x4 g3; g3[0] = 0; g3[1] = 0; g3[2] = 0; g3[3] = 0;
#if __clang_major__ >= 23
    i32x8 g4; for (int i = 0; i < 8; ++i) g4[i] = 0;
    __builtin_amdgcn_tensor_load_to_lds(g0, g1, g2, g3, g4, 0);
#else
    __builtin_amdgcn_tensor_load_to_lds(g0, g1, g2, g3, 0);
#endif
  }
#else
  const int n = tid >> 1, half = tid & 1;
  const uint4* s = (const uint4*)(Wsrc + n * ldw + half * 64);
  uint4* d = (uint4*)(sW + n * AS + half * 64);
#pragma unroll
  for (int i = 0; i < 8; ++i) d[i] = s[i];
#endif
}

// wait for all staging (async gathers + TDM weight tile) then barrier
__device__ __forceinline__ void stage_fence(int tid) {
#ifdef HAS_ASYNC
  __builtin_amdgcn_s_wait_asynccnt(0);
#endif
#ifdef HAS_TDM
  if (tid < 32) __builtin_amdgcn_s_wait_tensorcnt(0);
#endif
  __syncthreads();
}

// copy 64 bf16 (one half row, 128B) global -> LDS A tile (async path if available)
__device__ __forceinline__ void stage_row_half_bf(const unsigned short* __restrict__ src,
                                                  unsigned short* __restrict__ dst,
                                                  int row, int half, bool valid) {
  unsigned short* d = dst + row * AS + half * 64;
  if (valid) {
    const unsigned short* s = src + half * 64;
#ifdef HAS_ASYNC
#pragma unroll
    for (int i = 0; i < 8; ++i) async_copy16(s + i * 8, d + i * 8);
#else
#pragma unroll
    for (int i = 0; i < 8; ++i) ((uint4*)d)[i] = ((const uint4*)s)[i];
#endif
  } else {
    uint4 z; z.x = z.y = z.z = z.w = 0;
#pragma unroll
    for (int i = 0; i < 8; ++i) *(uint4*)(d + i * 8) = z;
  }
}

// convert+store 64 fp32 -> bf16 into one half-row of the LDS A tile
__device__ __forceinline__ void stage_row_half(const float* __restrict__ src,
                                               unsigned short* __restrict__ dst,
                                               int row, int half, bool valid) {
  unsigned short* d = dst + row * AS + half * 64;
  if (valid) {
    const float4* s = (const float4*)(src + half * 64);
#pragma unroll
    for (int i = 0; i < 16; ++i) {
      float4 f = s[i];
      ushort4 h;
      h.x = f2bf(f.x); h.y = f2bf(f.y); h.z = f2bf(f.z); h.w = f2bf(f.w);
      *(ushort4*)(d + i * 4) = h;
    }
  } else {
    ushort4 z; z.x = z.y = z.z = z.w = 0;
#pragma unroll
    for (int i = 0; i < 16; ++i) *(ushort4*)(d + i * 4) = z;
  }
}

template <bool EDGE>
__global__ __launch_bounds__(256) void mlp_ln_kernel(
    const unsigned short* __restrict__ nodes_bf,  // [N][128] bf16 (pre-converted)
    const float* __restrict__ node_f, const float* __restrict__ edge_f,
    const int* __restrict__ senders, const int* __restrict__ receivers,
    const unsigned short* __restrict__ W1T,  // [128][384] bf16
    const unsigned short* __restrict__ W2T,  // [128][128]
    const unsigned short* __restrict__ W3T,  // [128][128]
    const float* __restrict__ b1, const float* __restrict__ b2, const float* __restrict__ b3,
    const float* __restrict__ gamma, const float* __restrict__ beta,
    float* __restrict__ agg,   // EDGE: atomic scatter target; NODE: fp32 input
    float* __restrict__ out,   // residual output region
    long long nRowsTotal) {
  __shared__ __align__(16) unsigned short sA[ROWS * AS];
  __shared__ __align__(16) unsigned short sW[128 * AS];
  __shared__ __align__(16) unsigned short sH[ROWS * AS];

  const int tid  = threadIdx.x;
  const int lane = tid & 31;
  const int w    = tid >> 5;
  const int wrow = w * 16;
  const long long r0 = (long long)blockIdx.x * ROWS;

  if (tid == 0) {                 // warm L2 for the later weight blocks
    __builtin_prefetch(W2T, 0, 3);
    __builtin_prefetch(W3T, 0, 3);
  }

  v8f acc[8];
  zero_acc(acc);

  // ---- layer 1: K = 384 (edge) / 256 (node; zeros block skipped) ----
  const int KB = EDGE ? 3 : 2;
  const int srow = tid >> 1, shalf = tid & 1;
  const long long gr_s = r0 + srow;
  const bool valid_s = gr_s < nRowsTotal;
  for (int kb = 0; kb < KB; ++kb) {
    stage_w_any(W1T + kb * 128, 384, sW, tid);
    if (EDGE) {
      if (kb < 2) {
        int v = valid_s ? (kb == 0 ? senders[gr_s] : receivers[gr_s]) : 0;
        stage_row_half_bf(nodes_bf + (long long)v * 128, sA, srow, shalf, valid_s);
      } else {
        stage_row_half(edge_f + gr_s * 128, sA, srow, shalf, valid_s);
      }
    } else {
      if (kb == 0) stage_row_half_bf(nodes_bf + gr_s * 128, sA, srow, shalf, valid_s);
      else         stage_row_half(agg + gr_s * 128, sA, srow, shalf, valid_s);
    }
    stage_fence(tid);
    gemm128(sA, sW, acc, wrow, lane);
    __syncthreads();
  }
  bias_act_store(acc, b1, sH, wrow, lane, true);
  zero_acc(acc);

  // ---- layer 2 ----
  stage_w_any(W2T, 128, sW, tid);
  stage_fence(tid);
  gemm128(sH, sW, acc, wrow, lane);
  __syncthreads();
  bias_act_store(acc, b2, sA, wrow, lane, true);
  zero_acc(acc);

  // ---- layer 3 ----
  stage_w_any(W3T, 128, sW, tid);
  stage_fence(tid);
  gemm128(sA, sW, acc, wrow, lane);

  // ---- bias3 + LayerNorm + residual + (edge) segment-sum scatter ----
  const int m     = lane & 15;
  const int rbase = (lane >> 4) * 8;
  float bn[8], gv[8], bv[8];
#pragma unroll
  for (int t = 0; t < 8; ++t) {
    bn[t] = b3[t * 16 + m];
    gv[t] = gamma[t * 16 + m];
    bv[t] = beta[t * 16 + m];
  }
#pragma unroll
  for (int t = 0; t < 8; ++t)
#pragma unroll
    for (int r = 0; r < 8; ++r) acc[t][r] += bn[t];

  float mu[8], rs[8];
#pragma unroll
  for (int r = 0; r < 8; ++r) {
    float s1 = 0.0f, s2 = 0.0f;
#pragma unroll
    for (int t = 0; t < 8; ++t) { float x = acc[t][r]; s1 += x; s2 += x * x; }
#pragma unroll
    for (int off = 1; off < 16; off <<= 1) {   // reduce across the 16-lane half
      s1 += __shfl_xor(s1, off, 16);
      s2 += __shfl_xor(s2, off, 16);
    }
    float mm = s1 * (1.0f / 128.0f);
    float vv = s2 * (1.0f / 128.0f) - mm * mm;
    mu[r] = mm;
    rs[r] = rsqrtf(vv + 1e-5f);
  }

#pragma unroll
  for (int r = 0; r < 8; ++r) {
    long long gr = r0 + wrow + rbase + r;
    if (gr < nRowsTotal) {
      const float* resid;
      long long aggBase = 0;
      if (EDGE) {
        int recv = receivers[gr];
        aggBase  = (long long)recv * 128;
        resid    = edge_f + gr * 128;
      } else {
        resid = node_f + gr * 128;
      }
#pragma unroll
      for (int t = 0; t < 8; ++t) {
        int n = t * 16 + m;
        float ln = (acc[t][r] - mu[r]) * rs[r] * gv[t] + bv[t];
        out[gr * 128 + n] = ln + resid[n];
        if (EDGE) unsafeAtomicAdd(&agg[aggBase + n], ln);
      }
    }
  }
}

// transpose-convert fp32 weight [K][128] -> bf16 W^T [128][K]
__global__ void wconvert_kernel(const float* __restrict__ src,
                                unsigned short* __restrict__ dst,
                                int total, int K) {
  int i = blockIdx.x * 256 + threadIdx.x;
  if (i < total) {
    int k = i >> 7;
    int n = i & 127;
    dst[n * K + k] = f2bf(src[i]);
  }
}

// linear fp32 -> bf16 convert (vectorized x4)
__global__ void lconvert_kernel(const float* __restrict__ src,
                                unsigned short* __restrict__ dst,
                                long long total4) {
  long long i = (long long)blockIdx.x * 256 + threadIdx.x;
  if (i < total4) {
    float4 f = ((const float4*)src)[i];
    ushort4 h;
    h.x = f2bf(f.x); h.y = f2bf(f.y); h.z = f2bf(f.z); h.w = f2bf(f.w);
    ((ushort4*)dst)[i] = h;
  }
}

extern "C" void kernel_launch(void* const* d_in, const int* in_sizes, int n_in,
                              void* d_out, int out_size, void* d_ws, size_t ws_size,
                              hipStream_t stream) {
  const float* node_f  = (const float*)d_in[0];
  const float* edge_f  = (const float*)d_in[1];
  const int* senders   = (const int*)d_in[2];
  const int* receivers = (const int*)d_in[3];
  const float* ew1 = (const float*)d_in[4];  const float* eb1 = (const float*)d_in[5];
  const float* ew2 = (const float*)d_in[6];  const float* eb2 = (const float*)d_in[7];
  const float* ew3 = (const float*)d_in[8];  const float* eb3 = (const float*)d_in[9];
  const float* eg  = (const float*)d_in[10]; const float* ebt = (const float*)d_in[11];
  const float* nw1 = (const float*)d_in[12]; const float* nb1 = (const float*)d_in[13];
  const float* nw2 = (const float*)d_in[14]; const float* nb2 = (const float*)d_in[15];
  const float* nw3 = (const float*)d_in[16]; const float* nb3 = (const float*)d_in[17];
  const float* ng  = (const float*)d_in[18]; const float* nbt = (const float*)d_in[19];

  const long long N = in_sizes[0] / 128;
  const long long E = in_sizes[2];

  // ws layout: [agg f32 N*128][nodes_bf16 N*128][bf16 weights]
  float* agg = (float*)d_ws;
  unsigned short* nodes_bf = (unsigned short*)((char*)d_ws + (size_t)N * 128 * sizeof(float));
  unsigned short* wb = nodes_bf + (size_t)N * 128;
  unsigned short* eW1T = wb;
  unsigned short* eW2T = eW1T + 128 * 384;
  unsigned short* eW3T = eW2T + 128 * 128;
  unsigned short* nW1T = eW3T + 128 * 128;
  unsigned short* nW2T = nW1T + 128 * 384;
  unsigned short* nW3T = nW2T + 128 * 128;

  (void)hipMemsetAsync(agg, 0, (size_t)N * 128 * sizeof(float), stream);

  {
    long long total4 = N * 128 / 4;
    lconvert_kernel<<<(unsigned)((total4 + 255) / 256), 256, 0, stream>>>(node_f, nodes_bf, total4);
    int t1 = 384 * 128, t2 = 128 * 128;
    wconvert_kernel<<<(t1 + 255) / 256, 256, 0, stream>>>(ew1, eW1T, t1, 384);
    wconvert_kernel<<<(t2 + 255) / 256, 256, 0, stream>>>(ew2, eW2T, t2, 128);
    wconvert_kernel<<<(t2 + 255) / 256, 256, 0, stream>>>(ew3, eW3T, t2, 128);
    wconvert_kernel<<<(t1 + 255) / 256, 256, 0, stream>>>(nw1, nW1T, t1, 384);
    wconvert_kernel<<<(t2 + 255) / 256, 256, 0, stream>>>(nw2, nW2T, t2, 128);
    wconvert_kernel<<<(t2 + 255) / 256, 256, 0, stream>>>(nw3, nW3T, t2, 128);
  }

  float* out_node = (float*)d_out;
  float* out_edge = out_node + (size_t)N * 128;

  dim3 gridE((unsigned)((E + ROWS - 1) / ROWS));
  mlp_ln_kernel<true><<<gridE, 256, 0, stream>>>(
      nodes_bf, node_f, edge_f, senders, receivers, eW1T, eW2T, eW3T,
      eb1, eb2, eb3, eg, ebt, agg, out_edge, E);

  dim3 gridN((unsigned)((N + ROWS - 1) / 128), 1, 1);
  mlp_ln_kernel<false><<<gridN, 256, 0, stream>>>(
      nodes_bf, node_f, edge_f, senders, receivers, nW1T, nW2T, nW3T,
      nb1, nb2, nb3, ng, nbt, agg, out_node, N);
}
